// RoutingBlock_17901423690025
// MI455X (gfx1250) — matrix operand order repaired and checked by reading, not compiled, not run on hardware
//
#include <hip/hip_runtime.h>
#include <hip/hip_bf16.h>

#define N_ROWS 32768
#define DIM    1024
#define M_EXP  16
#define TOPK   4

typedef __attribute__((ext_vector_type(16))) __bf16 v16bf;
typedef __attribute__((ext_vector_type(8)))  float  v8f;

__device__ __forceinline__ void cvt4(v16bf& r, int base, float4 v) {
    r[base + 0] = (__bf16)v.x;
    r[base + 1] = (__bf16)v.y;
    r[base + 2] = (__bf16)v.z;
    r[base + 3] = (__bf16)v.w;
}

// A fragment (16-bit A 16x32, ISA 7.12.2): lane holds row (lane&15);
// elements 0..15 -> K = kb + half*16 + e (contiguous)
// p already includes row base + half*16.
__device__ __forceinline__ v16bf load_frag_a(const float* __restrict__ p) {
    v16bf r;
    cvt4(r, 0,  *(const float4*)(p));
    cvt4(r, 4,  *(const float4*)(p + 4));
    cvt4(r, 8,  *(const float4*)(p + 8));
    cvt4(r, 12, *(const float4*)(p + 12));
    return r;
}

// B fragment (16-bit B 32x16): lane holds column (lane&15);
// elements 0..15 -> K = kb + half*16 + e (contiguous). p includes row base + half*16.
__device__ __forceinline__ v16bf load_frag_b(const float* __restrict__ p) {
    v16bf r;
    cvt4(r, 0,  *(const float4*)(p));
    cvt4(r, 4,  *(const float4*)(p + 4));
    cvt4(r, 8,  *(const float4*)(p + 8));
    cvt4(r, 12, *(const float4*)(p + 12));
    return r;
}

__global__ __launch_bounds__(256)
void routing_topk_kernel(const float* __restrict__ x,
                         const float* __restrict__ noise,
                         const float* __restrict__ W_r,
                         const float* __restrict__ b_r,
                         const float* __restrict__ W_noise,
                         const float* __restrict__ b_noise,
                         float* __restrict__ out) {
    const int lane = threadIdx.x & 31;
    const int wave = threadIdx.x >> 5;
    const int tile = blockIdx.x * (blockDim.x >> 5) + wave;
    const int row0 = tile * 16;

    const int e    = lane & 15;   // expert column (and A row within tile)
    const int half = lane >> 4;

    const float* __restrict__ xp  = x       + (row0 + e) * DIM + half * 16;  // A layout
    const float* __restrict__ wrp = W_r     + e * DIM + half * 16;           // B layout
    const float* __restrict__ wnp = W_noise + e * DIM + half * 16;

    v8f acc_r = {0.f, 0.f, 0.f, 0.f, 0.f, 0.f, 0.f, 0.f};
    v8f acc_n = {0.f, 0.f, 0.f, 0.f, 0.f, 0.f, 0.f, 0.f};

    #pragma unroll 2
    for (int kb = 0; kb < DIM; kb += 32) {
        if (kb + 32 < DIM) {
            __builtin_prefetch(xp + kb + 32, 0, 1);   // global_prefetch_b8, next K-block
        }
        v16bf a  = load_frag_a(xp  + kb);
        v16bf br = load_frag_b(wrp + kb);
        v16bf bn = load_frag_b(wnp + kb);
        acc_r = __builtin_amdgcn_wmma_f32_16x16x32_bf16(
                    false, a, false, br, (short)0, acc_r, false, false);
        acc_n = __builtin_amdgcn_wmma_f32_16x16x32_bf16(
                    false, a, false, bn, (short)0, acc_n, false, false);
    }

    const float brs = b_r[e];
    const float bns = b_noise[e];

    // C/D layout: lane holds N = lane&15 (expert); VGPR r holds row M = r + 8*half.
    #pragma unroll
    for (int r = 0; r < 8; ++r) {
        const int n = row0 + r + 8 * half;

        const float score = acc_r[r] + brs;
        const float nb    = acc_n[r] + bns;
        // numerically stable softplus
        const float sp    = fmaxf(nb, 0.0f) + log1pf(__expf(-fabsf(nb)));
        const float raw   = fmaf(noise[n * M_EXP + e], sp, score);

        // softmax over 16 experts: butterfly within the 16-lane half-group
        float mx = raw;
        mx = fmaxf(mx, __shfl_xor(mx, 1, 32));
        mx = fmaxf(mx, __shfl_xor(mx, 2, 32));
        mx = fmaxf(mx, __shfl_xor(mx, 4, 32));
        mx = fmaxf(mx, __shfl_xor(mx, 8, 32));
        const float ex = __expf(raw - mx);
        float s = ex;
        s += __shfl_xor(s, 1, 32);
        s += __shfl_xor(s, 2, 32);
        s += __shfl_xor(s, 4, 32);
        s += __shfl_xor(s, 8, 32);
        const float soft = ex / s;

        // top-4: rank = #peers strictly better (ties broken by lower expert index)
        int rank = 0;
        #pragma unroll
        for (int i = 1; i < 16; ++i) {
            const float pv = __shfl_xor(soft, i, 32);
            const int pidx = e ^ i;
            rank += (pv > soft) || ((pv == soft) && (pidx < e));
        }

        out[n * M_EXP + e] = (rank < TOPK) ? soft : 0.0f;
    }
}

extern "C" void kernel_launch(void* const* d_in, const int* in_sizes, int n_in,
                              void* d_out, int out_size, void* d_ws, size_t ws_size,
                              hipStream_t stream) {
    const float* x       = (const float*)d_in[0];
    const float* noise   = (const float*)d_in[1];
    const float* W_r     = (const float*)d_in[2];
    const float* b_r     = (const float*)d_in[3];
    const float* W_noise = (const float*)d_in[4];
    const float* b_noise = (const float*)d_in[5];
    float* out = (float*)d_out;

    const int waves_per_block = 8;                       // 256 threads
    const int tiles = N_ROWS / 16;                       // 2048 wave-tiles
    dim3 grid(tiles / waves_per_block);                  // 256 blocks
    dim3 block(32 * waves_per_block);

    hipLaunchKernelGGL(routing_topk_kernel, grid, block, 0, stream,
                       x, noise, W_r, b_r, W_noise, b_noise, out);
}
